// MultiHeadAttention_24653112279652
// MI455X (gfx1250) — matrix operand order repaired and checked by reading, not compiled, not run on hardware
//
#include <hip/hip_runtime.h>
#include <hip/hip_bf16.h>

// ---------------------------------------------------------------------------
// Problem constants (fixed by the reference)
// B=2, T=2048, C=1024, H=16, dk=dv=64
// M = B*T = 4096 rows for all projection GEMMs, N = K = 1024.
// ---------------------------------------------------------------------------

typedef __attribute__((ext_vector_type(16))) __bf16       v16bf;
typedef __attribute__((ext_vector_type(8)))  float        v8f;
typedef __attribute__((ext_vector_type(4)))  float        f32x4;
typedef __attribute__((ext_vector_type(4)))  unsigned int u32x4;
// TDM descriptor groups (6-arg clang-23 builtin form)
typedef __attribute__((ext_vector_type(4))) unsigned int tdm_g0_t;
typedef __attribute__((ext_vector_type(8))) int          tdm_g1_t;
typedef __attribute__((ext_vector_type(4))) int          tdm_g2_t;

union Frag16 {            // one 16x32 A-frag or 32x16 B-frag of bf16 (8 VGPRs)
    u32x4  q[2];
    v16bf  v;
    __bf16 e[16];
};

__device__ __forceinline__ __bf16 f2bf(float f) {
    union { float f; unsigned u; } in; in.f = f;
    unsigned r = (in.u + 0x7FFFu + ((in.u >> 16) & 1u)) >> 16;   // RNE
    union { unsigned short s; __bf16 b; } out; out.s = (unsigned short)r;
    return out.b;
}

__device__ __forceinline__ v8f wmma_bf16(v16bf a, v16bf b, v8f c) {
    // D = A(16x32 bf16) * B(32x16 bf16) + C(16x16 f32)
    return __builtin_amdgcn_wmma_f32_16x16x32_bf16(
        /*neg_a=*/false, a, /*neg_b=*/false, b,
        /*c_mod=*/(short)0, c, /*reuse_a=*/false, /*reuse_b=*/false);
}

// ---------------------------------------------------------------------------
// Tiled GEMM:  Out(4096 x 1024) = A(4096 x 1024) @ W(1024 x 1024) + bias
//   AMODE 0: A is fp32 (raw Q/K/V input), 1: A is bf16 (attention output)
//   OMODE 0: fp32 plain row-major output (final projection)
//   OMODE 1: bf16 output scattered to (B,H,T,64) head layout (q/k/v)
// Block = 256 threads (8 waves), block tile 128x128, K-step 32.
// Each wave owns a 64x32 subtile = 4x2 WMMA accumulators.
// ---------------------------------------------------------------------------
template <int AMODE, int OMODE>
__global__ __launch_bounds__(256)
void gemm_bf16_kernel(const void* __restrict__ Aptr,
                      const float* __restrict__ W,
                      const float* __restrict__ bias,
                      void* __restrict__ Out,
                      float scale)
{
    constexpr int Kd = 1024, Nn = 1024;

    // LDS tiles, padded so rows stay 16B-aligned and banks are spread.
    __shared__ __bf16 As[128][40];   // 128x32 used, stride 40 (80B, 16B-aligned)
    __shared__ __bf16 Bs[32][136];   // 32x128 used, stride 136 (272B, 16B-aligned)

    const int tid    = threadIdx.x;
    const int lane   = tid & 31;
    const int wave   = tid >> 5;
    const int half   = lane >> 4;     // 0 / 1
    const int lanelo = lane & 15;
    const int wm     = (wave & 1) * 64;   // wave row offset in block tile
    const int wn     = (wave >> 1) * 32;  // wave col offset in block tile
    const int mblk   = blockIdx.y * 128;
    const int nblk   = blockIdx.x * 128;

    v8f acc[4][2];
    #pragma unroll
    for (int i = 0; i < 4; ++i)
        #pragma unroll
        for (int j = 0; j < 2; ++j)
            #pragma unroll
            for (int r = 0; r < 8; ++r) acc[i][j][r] = 0.0f;

    const int arow = tid >> 1, acol = (tid & 1) * 16;   // A tile: 128 rows x 32
    const int brow = tid >> 3, bcol = (tid & 7) * 16;   // B tile: 32 rows x 128

    for (int k0 = 0; k0 < Kd; k0 += 32) {
        // ---- stage A tile (convert fp32 -> bf16 if needed) ----
        if (AMODE == 0) {
            const float* ap = (const float*)Aptr + (size_t)(mblk + arow) * Kd + k0 + acol;
            #pragma unroll
            for (int q4 = 0; q4 < 4; ++q4) {
                f32x4 t = *(const f32x4*)(ap + q4 * 4);
                #pragma unroll
                for (int i = 0; i < 4; ++i) As[arow][acol + q4 * 4 + i] = f2bf(t[i]);
            }
            if (k0 + 32 < Kd) __builtin_prefetch(ap + 32, 0, 0);
        } else {
            const __bf16* ap = (const __bf16*)Aptr + (size_t)(mblk + arow) * Kd + k0 + acol;
            *(u32x4*)&As[arow][acol]     = *(const u32x4*)ap;
            *(u32x4*)&As[arow][acol + 8] = *(const u32x4*)(ap + 8);
            if (k0 + 32 < Kd) __builtin_prefetch(ap + 32, 0, 0);
        }
        // ---- stage B tile (weights fp32 -> bf16) ----
        {
            const float* bp = W + (size_t)(k0 + brow) * Nn + nblk + bcol;
            #pragma unroll
            for (int q4 = 0; q4 < 4; ++q4) {
                f32x4 t = *(const f32x4*)(bp + q4 * 4);
                #pragma unroll
                for (int i = 0; i < 4; ++i) Bs[brow][bcol + q4 * 4 + i] = f2bf(t[i]);
            }
            if (k0 + 32 < Kd) __builtin_prefetch(bp + (size_t)32 * Nn, 0, 0);
        }
        __syncthreads();

        // ---- fragments from LDS (ISA 16-bit A/B VGPR layouts) ----
        Frag16 afr[4], bfr[2];
        #pragma unroll
        for (int i = 0; i < 4; ++i) {
            // A 16x32: lane -> row m, regs 0-3 = K[half*8 .. +7], regs 4-7 = +16
            const __bf16* p = &As[wm + i * 16 + lanelo][half * 8];
            afr[i].q[0] = *(const u32x4*)p;
            afr[i].q[1] = *(const u32x4*)(p + 16);
        }
        #pragma unroll
        for (int j = 0; j < 2; ++j) {
            // B 32x16: lane -> column n, regs 0-3 = K[half*8 .. +7], regs 4-7 = +16
            #pragma unroll
            for (int s = 0; s < 16; ++s)
                bfr[j].e[s] = Bs[(s & 7) + half * 8 + (s >> 3) * 16][wn + j * 16 + lanelo];
        }
        #pragma unroll
        for (int i = 0; i < 4; ++i)
            #pragma unroll
            for (int j = 0; j < 2; ++j)
                acc[i][j] = wmma_bf16(afr[i].v, bfr[j].v, acc[i][j]);
        __syncthreads();
    }

    // ---- epilogue: bias (+ optional scale), scatter ----
    #pragma unroll
    for (int j = 0; j < 2; ++j) {
        const int ng = nblk + wn + j * 16 + lanelo;
        const float bj = bias[ng];
        #pragma unroll
        for (int i = 0; i < 4; ++i) {
            #pragma unroll
            for (int r = 0; r < 8; ++r) {
                const int mg = mblk + wm + i * 16 + r + half * 8;  // C layout: M = r + 8*half
                const float val = (acc[i][j][r] + bj) * scale;
                if (OMODE == 0) {
                    ((float*)Out)[(size_t)mg * Nn + ng] = val;
                } else {
                    const int bb = mg >> 11, tt = mg & 2047;   // T = 2048
                    const int hh = ng >> 6,  dd = ng & 63;     // d = 64
                    ((__bf16*)Out)[(((size_t)(bb * 16 + hh)) * 2048 + tt) * 64 + dd] = f2bf(val);
                }
            }
        }
    }
}

// ---------------------------------------------------------------------------
// Flash attention: one block per (b, h, 128 q-rows); 8 waves x 16 q-rows.
// Loop over 64-key tiles, online softmax, output accumulated in fp32 WMMA
// accumulators, written as bf16 in (B, T, H*64) layout for the final GEMM.
// q was pre-scaled by 1/sqrt(64) in the projection epilogue.
// V tile staging is offloaded to the Tensor Data Mover (tensor_load_to_lds,
// tracked with TENSORcnt) so the vector pipe stays free for WMMA issue.
// ---------------------------------------------------------------------------
__global__ __launch_bounds__(256)
void flash_attn_kernel(const __bf16* __restrict__ qg,
                       const __bf16* __restrict__ kg,
                       const __bf16* __restrict__ vg,
                       __bf16* __restrict__ og)
{
    constexpr int T = 2048, D = 64, KT = 64;

    __shared__ __bf16 Kt[64][64];      // K^T tile: [qk-dim][key]      (8 KB)
    __shared__ __bf16 Vs[64][64];      // V tile:   [key][v-dim]       (8 KB)
    __shared__ __bf16 Ps[8][16][64];   // per-wave P staging           (16 KB)

    const int tid    = threadIdx.x;
    const int lane   = tid & 31;
    const int wave   = tid >> 5;
    const int half   = lane >> 4;
    const int lanelo = lane & 15;

    const int qt = blockIdx.x & 15;          // q tile (128 rows each)
    const int h  = (blockIdx.x >> 4) & 15;
    const int b  = blockIdx.x >> 8;
    const size_t base = ((size_t)(b * 16 + h)) * T * D;

    // Load this wave's Q fragments (2 k-chunks of 32) straight from global.
    Frag16 aq[2];
    {
        const int tq = qt * 128 + wave * 16 + lanelo;      // A layout: lane -> row
        const __bf16* qrow = qg + base + (size_t)tq * D;
        #pragma unroll
        for (int c = 0; c < 2; ++c) {
            aq[c].q[0] = *(const u32x4*)(qrow + c * 32 + half * 8);
            aq[c].q[1] = *(const u32x4*)(qrow + c * 32 + 16 + half * 8);
        }
    }

    // ---- TDM descriptor, invariant parts (ISA 8.3/8.4 D# layout) ----
    // V tile: 64 rows x 64 bf16, row stride 64 elements, inside a 2048-row tensor.
    const unsigned lds_v = (unsigned)(size_t)&Vs[0][0];   // LDS byte offset
    tdm_g1_t g1;
    g1[0] = (1 << 16);            // wg_mask=0, data_size=1 (2 bytes)
    g1[1] = (64 << 16);           // tensor_dim0[15:0]=64 at bits 79:48
    g1[2] = (2048 << 16);         // dim0 hi=0; tensor_dim1[15:0]=2048 at bits 111:80
    g1[3] = (64 << 16);           // dim1 hi=0; tile_dim0=64 at bits 127:112
    g1[4] = 64;                   // tile_dim1=64 (bits 143:128), tile_dim2=0
    g1[5] = 64;                   // tensor_dim0_stride=64 (bits 207:160, low dword)
    g1[6] = 0;                    // stride hi, tensor_dim1_stride lo
    g1[7] = 0;
    const tdm_g2_t gz  = {0, 0, 0, 0};           // 2D tensor: groups 2/3 unused
    const tdm_g1_t gz8 = {0, 0, 0, 0, 0, 0, 0, 0};  // trailing group (clang-23 form)

    float runM[8], runL[8];
    v8f   o[4];
    #pragma unroll
    for (int r = 0; r < 8; ++r) { runM[r] = -1e30f; runL[r] = 0.0f; }
    #pragma unroll
    for (int vj = 0; vj < 4; ++vj)
        #pragma unroll
        for (int r = 0; r < 8; ++r) o[vj][r] = 0.0f;

    const int kr   = tid & 63;     // key row this thread stages
    const int part = tid >> 6;     // 16-element dim slice

    for (int kt = 0; kt < T / KT; ++kt) {
        const int key0 = kt * KT;
        __syncthreads();   // previous tile fully consumed before overwrite

        // ---- V tile via Tensor Data Mover (wave 0 only, uniform D#) ----
        if (wave == 0) {
            const unsigned long long ga =
                (unsigned long long)(const void*)(vg + base + (size_t)key0 * D);
            tdm_g0_t g0;
            g0[0] = 1u;                                      // count=1, user mode
            g0[1] = lds_v;                                   // lds_addr
            g0[2] = (unsigned)ga;                            // global_addr[31:0]
            g0[3] = (unsigned)((ga >> 32) & 0x1FFFFFFu)      // global_addr[56:32]
                    | (2u << 30);                            // type=2 ("image")
            __builtin_amdgcn_tensor_load_to_lds(g0, g1, gz, gz, gz8, 0);
        }

        // Stage K transposed (so QK^T B-frags are contiguous row reads).
        {
            const __bf16* krow = kg + base + (size_t)(key0 + kr) * D + part * 16;
            Frag16 tv;
            tv.q[0] = *(const u32x4*)krow;
            tv.q[1] = *(const u32x4*)(krow + 8);
            #pragma unroll
            for (int i = 0; i < 16; ++i) Kt[part * 16 + i][kr] = tv.e[i];
        }

        if (wave == 0) __builtin_amdgcn_s_wait_tensorcnt(0);  // V tile landed
        __syncthreads();

        // ---- S = q @ K^T  (16 x 64 per wave, fp32) ----
        v8f s[4];
        #pragma unroll
        for (int j = 0; j < 4; ++j) {
            #pragma unroll
            for (int r = 0; r < 8; ++r) s[j][r] = 0.0f;
            #pragma unroll
            for (int c = 0; c < 2; ++c) {
                Frag16 bk;
                #pragma unroll
                for (int e = 0; e < 16; ++e)   // B: lane -> n column
                    bk.e[e] = Kt[c * 32 + (e & 7) + half * 8 + (e >> 3) * 16][j * 16 + lanelo];
                s[j] = wmma_bf16(aq[c].v, bk.v, s[j]);
            }
        }

        // ---- online softmax (rows of C layout: M = r + 8*half) ----
        float corr[8], rs[8];
        #pragma unroll
        for (int r = 0; r < 8; ++r) {
            float m = s[0][r];
            #pragma unroll
            for (int j = 1; j < 4; ++j) m = fmaxf(m, s[j][r]);
            #pragma unroll
            for (int off = 1; off < 16; off <<= 1)       // stays inside 16-lane half
                m = fmaxf(m, __shfl_xor(m, off, 32));
            const float nm = fmaxf(runM[r], m);
            corr[r] = __expf(runM[r] - nm);
            runM[r] = nm;
            rs[r]   = 0.0f;
        }
        #pragma unroll
        for (int j = 0; j < 4; ++j) {
            #pragma unroll
            for (int r = 0; r < 8; ++r) {
                const float p = __expf(s[j][r] - runM[r]);
                rs[r] += p;
                Ps[wave][r + half * 8][j * 16 + lanelo] = f2bf(p);   // C -> row-major
            }
        }
        #pragma unroll
        for (int r = 0; r < 8; ++r) {
            float t = rs[r];
            #pragma unroll
            for (int off = 1; off < 16; off <<= 1) t += __shfl_xor(t, off, 32);
            runL[r] = runL[r] * corr[r] + t;
        }
        #pragma unroll
        for (int vj = 0; vj < 4; ++vj)
            #pragma unroll
            for (int r = 0; r < 8; ++r) o[vj][r] *= corr[r];

        // ---- reload P as A-fragments, O += P @ V ----
        Frag16 ap[2];
        #pragma unroll
        for (int c = 0; c < 2; ++c) {
            const __bf16* pp = &Ps[wave][lanelo][c * 32 + half * 8];
            ap[c].q[0] = *(const u32x4*)pp;
            ap[c].q[1] = *(const u32x4*)(pp + 16);
        }
        #pragma unroll
        for (int vj = 0; vj < 4; ++vj) {
            #pragma unroll
            for (int c = 0; c < 2; ++c) {
                Frag16 bv;
                #pragma unroll
                for (int e = 0; e < 16; ++e)
                    bv.e[e] = Vs[c * 32 + (e & 7) + half * 8 + (e >> 3) * 16][vj * 16 + lanelo];
                o[vj] = wmma_bf16(ap[c].v, bv.v, o[vj]);
            }
        }
    }

    // ---- normalize, write (B, T, H*64) bf16 for the output projection ----
    #pragma unroll
    for (int vj = 0; vj < 4; ++vj) {
        #pragma unroll
        for (int r = 0; r < 8; ++r) {
            const int t  = qt * 128 + wave * 16 + r + half * 8;
            const int vd = vj * 16 + lanelo;
            const float val = o[vj][r] / runL[r];
            og[((size_t)(b * T + t)) * 1024 + h * 64 + vd] = f2bf(val);
        }
    }
}

// ---------------------------------------------------------------------------
// Host launcher
// ---------------------------------------------------------------------------
extern "C" void kernel_launch(void* const* d_in, const int* in_sizes, int n_in,
                              void* d_out, int out_size, void* d_ws, size_t ws_size,
                              hipStream_t stream) {
    (void)in_sizes; (void)n_in; (void)out_size; (void)ws_size;

    const float* Q  = (const float*)d_in[0];
    const float* Kx = (const float*)d_in[1];
    const float* V  = (const float*)d_in[2];
    const float* Wq = (const float*)d_in[3];
    const float* bq = (const float*)d_in[4];
    const float* bk = (const float*)d_in[6];
    const float* Wk = (const float*)d_in[5];
    const float* Wv = (const float*)d_in[7];
    const float* bv = (const float*)d_in[8];
    const float* Wo = (const float*)d_in[9];
    const float* bo = (const float*)d_in[10];

    // Workspace: 4 x (B*T*C) bf16 buffers = 4 x 8 MB
    const size_t E = (size_t)2 * 2048 * 1024;
    __bf16* qw = (__bf16*)d_ws;
    __bf16* kw = qw + E;
    __bf16* vw = kw + E;
    __bf16* aw = vw + E;

    dim3 grid(1024 / 128, 4096 / 128);   // (N tiles, M tiles) = (8, 32)
    dim3 block(256);

    // Q/K/V projections -> bf16 head layout; softmax scale folded into q.
    gemm_bf16_kernel<0, 1><<<grid, block, 0, stream>>>(Q,  Wq, bq, qw, 0.125f);
    gemm_bf16_kernel<0, 1><<<grid, block, 0, stream>>>(Kx, Wk, bk, kw, 1.0f);
    gemm_bf16_kernel<0, 1><<<grid, block, 0, stream>>>(V,  Wv, bv, vw, 1.0f);

    // Flash attention: B*H*(T/128) = 2*16*16 = 512 blocks.
    flash_attn_kernel<<<dim3(512), block, 0, stream>>>(qw, kw, vw, aw);

    // Output projection -> fp32 d_out.
    gemm_bf16_kernel<1, 0><<<grid, block, 0, stream>>>(aw, Wo, bo, (float*)d_out, 1.0f);
}
